// Classifier_4604204941380
// MI455X (gfx1250) — matrix operand order repaired and checked by reading, not compiled, not run on hardware
//
#include <hip/hip_runtime.h>
#include <hip/hip_bf16.h>
#include <math.h>

// ---------------- WMMA types (CDNA5 / gfx1250, wave32) ----------------
typedef __attribute__((ext_vector_type(16))) _Float16 v16h;
typedef __attribute__((ext_vector_type(8)))  float    v8f;

#define F_IN  1433
#define D_HID 64
#define HEADS 4
#define DHEAD 16
#define DEPTH 3
#define ALPHA 0.9f
#define TK    32   // K-tile for WMMA GEMM
#define MT    32   // M rows per block

// monotone float <-> uint encoding so atomicMax(uint) == float max
__device__ __forceinline__ unsigned fenc(float f) {
    unsigned u = __float_as_uint(f);
    return (u & 0x80000000u) ? ~u : (u | 0x80000000u);
}
__device__ __forceinline__ float fdec(unsigned e) {
    unsigned u = (e & 0x80000000u) ? (e & 0x7FFFFFFFu) : ~e;
    return __uint_as_float(u);
}
#define ENC_NEG_INF 0x007FFFFFu   // fenc(-inf)

// ---------------- CDNA5 async Global->LDS DMA (ISA 10.7 / 15.18.3) ----------------
// IOFFSET applies to BOTH lds and global addresses -> one base, many offsets.
template <int OFF>
__device__ __forceinline__ void async_ld_b32_nt(unsigned lds, const void* g) {
    asm volatile("global_load_async_to_lds_b32 %0, %1, off offset:%2 th:TH_LOAD_NT"
                 :: "v"(lds), "v"(g), "i"(OFF) : "memory");
}
template <int OFF>
__device__ __forceinline__ void async_ld_b128(unsigned lds, const void* g) {
    asm volatile("global_load_async_to_lds_b128 %0, %1, off offset:%2"
                 :: "v"(lds), "v"(g), "i"(OFF) : "memory");
}
#define WAIT_ASYNC(n) asm volatile("s_wait_asynccnt %0" :: "i"(n) : "memory")

// flat LDS pointer -> 32-bit LDS offset (aperture occupies addr[63:32])
__device__ __forceinline__ unsigned lds_off(const void* p) {
    return (unsigned)(size_t)p;
}

// ---------------------------------------------------------------------
// GEMM: C[M,64] = A[M,K] * B[K,64] (+bias).  Block = 128 thr = 4 waves.
// Block tile: 32 rows x 64 cols; wave w owns cols [16w,16w+16) with TWO
// 16x16 accumulators (rows 0-15 / 16-31) sharing one B fragment.
// Tiles stream Global->LDS via async DMA, double-buffered software
// pipeline.  A16: A rows 16B-aligned (K%4==0).  Non-A16 path streams A
// non-temporally (573 MB stream vs 192 MB L2).
// Requires M % 32 == 0 (true: 100000 = 3125*32).
// ---------------------------------------------------------------------
template <bool A16>
__global__ void gemm_wmma_n64(const float* __restrict__ A,
                              const float* __restrict__ B,
                              const float* __restrict__ bias,
                              float* __restrict__ C,
                              int M, int K) {
    __shared__ float sA[2][MT * TK];    // 2 x 4 KB
    __shared__ float sB[2][TK * 64];    // 2 x 8 KB

    const int tid  = threadIdx.x;          // 0..127
    const int wave = tid >> 5;
    const int lane = tid & 31;
    const int half = lane >> 4;
    const int lr   = lane & 15;
    const int row0 = blockIdx.x * MT;
    const int n0   = wave * 16;

    // cooperative DMA coordinates: 8 A floats + 16 B floats per thread
    const int ar  = tid >> 2;              // 0..31  (A row)
    const int ak0 = (tid & 3) * 8;         // 0,8,16,24
    const int bk  = tid >> 2;              // 0..31  (B k)
    const int bn0 = (tid & 3) * 16;        // 0,16,32,48

    const int kfull = K / TK;
    const int krem  = K - kfull * TK;
    constexpr int INFLIGHT = A16 ? 6 : 12; // async ops per thread per tile

    v8f accLo = {}, accHi = {};

    auto issue_tile = [&](int kt, int buf) {
        const float* Ap = A + (size_t)(row0 + ar) * K + kt * TK + ak0;
        const float* Bp = B + (size_t)(kt * TK + bk) * 64 + bn0;
        const unsigned la = lds_off(&sA[buf][ar * TK + ak0]);
        const unsigned lb = lds_off(&sB[buf][bk * 64 + bn0]);
        if (A16) {
            async_ld_b128<0>(la, Ap);
            async_ld_b128<16>(la, Ap);
        } else {               // odd-K rows: 4B alignment only; NT stream
            async_ld_b32_nt<0>(la, Ap);
            async_ld_b32_nt<4>(la, Ap);
            async_ld_b32_nt<8>(la, Ap);
            async_ld_b32_nt<12>(la, Ap);
            async_ld_b32_nt<16>(la, Ap);
            async_ld_b32_nt<20>(la, Ap);
            async_ld_b32_nt<24>(la, Ap);
            async_ld_b32_nt<28>(la, Ap);
        }
        async_ld_b128<0>(lb, Bp);
        async_ld_b128<16>(lb, Bp);
        async_ld_b128<32>(lb, Bp);
        async_ld_b128<48>(lb, Bp);
    };

    auto compute_tile = [&](const float* __restrict__ cA, const float* __restrict__ cB) {
        v16h aLo, aHi, b;
#pragma unroll
        for (int e = 0; e < 8; ++e) {
            aLo[e]     = (_Float16)cA[lr * TK + half * 8 + e];
            aLo[e + 8] = (_Float16)cA[lr * TK + 16 + half * 8 + e];
            aHi[e]     = (_Float16)cA[(16 + lr) * TK + half * 8 + e];
            aHi[e + 8] = (_Float16)cA[(16 + lr) * TK + 16 + half * 8 + e];
        }
#pragma unroll
        for (int e = 0; e < 16; ++e)
            b[e] = (_Float16)cB[(half * 8 + (e & 7) + (e >> 3) * 16) * 64 + n0 + lr];
        accLo = __builtin_amdgcn_wmma_f32_16x16x32_f16(
                    false, aLo, false, b, (short)0, accLo, false, false);
        accHi = __builtin_amdgcn_wmma_f32_16x16x32_f16(
                    false, aHi, false, b, (short)0, accHi, false, false);
    };

    if (kfull > 0) {
        issue_tile(0, 0);                       // prologue
        for (int kt = 0; kt < kfull - 1; ++kt) { // steady state: branch-free wait
            issue_tile(kt + 1, (kt + 1) & 1);
            WAIT_ASYNC(INFLIGHT);               // tile kt landed (in-order per wave)
            if (kt + 2 < kfull)
                __builtin_prefetch(A + (size_t)(row0 + ar) * K + (kt + 2) * TK + ak0, 0, 1);
            __syncthreads();
            compute_tile(sA[kt & 1], sB[kt & 1]);
            __syncthreads();                    // readers done before next overwrite
        }
        WAIT_ASYNC(0);                          // epilogue: last tile
        __syncthreads();
        compute_tile(sA[(kfull - 1) & 1], sB[(kfull - 1) & 1]);
        __syncthreads();
    }

    if (krem) {                                 // guarded tail tile (plain loads)
        const int k0 = kfull * TK;
#pragma unroll
        for (int j = 0; j < 8; ++j) {
            const int gk = ak0 + j;
            sA[0][ar * TK + gk] = (gk < krem) ? A[(size_t)(row0 + ar) * K + k0 + gk] : 0.f;
        }
#pragma unroll
        for (int j = 0; j < 16; ++j)
            sB[0][bk * 64 + bn0 + j] = (bk < krem) ? B[(size_t)(k0 + bk) * 64 + bn0 + j] : 0.f;
        __syncthreads();
        compute_tile(sA[0], sB[0]);
    }

    // ---- store: C/D layout: lane lr = N col; VGPR j -> row half*8+j ----
    const float bv = bias ? bias[n0 + lr] : 0.f;
#pragma unroll
    for (int j = 0; j < 8; ++j) {
        const int r = half * 8 + j;
        C[(size_t)(row0 + r) * 64 + n0 + lr]      = accLo[j] + bv;
        C[(size_t)(row0 + 16 + r) * 64 + n0 + lr] = accHi[j] + bv;
    }
}

// ---------------- LayerNorm: one wave32 per node ----------------
__global__ void layernorm_wave(const float* __restrict__ x,
                               float* __restrict__ h,
                               const float* __restrict__ sc,
                               const float* __restrict__ bi, int N) {
    const int gid  = blockIdx.x * blockDim.x + threadIdx.x;
    const int node = gid >> 5;
    const int lane = threadIdx.x & 31;
    if (node >= N) return;
    const float* xp = x + (size_t)node * 64;
    float a = xp[lane], b = xp[lane + 32];
    float s = a + b, ss = a * a + b * b;
#pragma unroll
    for (int o = 16; o > 0; o >>= 1) {
        s  += __shfl_xor(s,  o, 32);
        ss += __shfl_xor(ss, o, 32);
    }
    const float mu  = s * (1.f / 64.f);
    const float var = ss * (1.f / 64.f) - mu * mu;
    const float r   = rsqrtf(var + 1e-5f);
    h[(size_t)node * 64 + lane]      = (a - mu) * r * sc[lane] + bi[lane];
    h[(size_t)node * 64 + lane + 32] = (b - mu) * r * sc[lane + 32] + bi[lane + 32];
}

// ---------------- per-layer accumulator init ----------------
__global__ void init_accum(float* __restrict__ agg, unsigned* __restrict__ menc,
                           float* __restrict__ denom, int N) {
    const int t = blockIdx.x * blockDim.x + threadIdx.x;
    if (t < N * 64) agg[t] = 0.f;
    if (t < N * HEADS) { menc[t] = ENC_NEG_INF; denom[t] = 0.f; }
}

// ---------------- edge logits + segment max ----------------
__global__ void edge_logits(const int* __restrict__ src, const int* __restrict__ dst,
                            const float* __restrict__ q, const float* __restrict__ k,
                            float* __restrict__ logits, unsigned* __restrict__ menc,
                            int E) {
    const int e = blockIdx.x * blockDim.x + threadIdx.x;
    if (e >= E) return;
    const int s = src[e], d = dst[e];
    const float* qp = q + (size_t)d * 64;
    const float* kp = k + (size_t)s * 64;
#pragma unroll
    for (int hh = 0; hh < HEADS; ++hh) {
        float acc = 0.f;
#pragma unroll
        for (int t = 0; t < DHEAD; ++t) acc += qp[hh * DHEAD + t] * kp[hh * DHEAD + t];
        acc *= 0.25f;   // DH^-0.5
        logits[(size_t)e * HEADS + hh] = acc;
        atomicMax(&menc[(size_t)d * HEADS + hh], fenc(acc));
    }
}

// ---------------- exp(logit - max) + segment sum ----------------
__global__ void edge_expsum(const int* __restrict__ dst,
                            float* __restrict__ logits,        // in: logits, out: exp
                            const unsigned* __restrict__ menc,
                            float* __restrict__ denom, int E) {
    const int e = blockIdx.x * blockDim.x + threadIdx.x;
    if (e >= E) return;
    const int d = dst[e];
#pragma unroll
    for (int hh = 0; hh < HEADS; ++hh) {
        const float m  = fdec(menc[(size_t)d * HEADS + hh]);
        const float ex = __expf(logits[(size_t)e * HEADS + hh] - m);
        logits[(size_t)e * HEADS + hh] = ex;
        atomicAdd(&denom[(size_t)d * HEADS + hh], ex);
    }
}

// ---------------- weighted scatter: agg[dst] += attn * v[src] ----------------
__global__ void edge_aggregate(const int* __restrict__ src, const int* __restrict__ dst,
                               const float* __restrict__ ex, const float* __restrict__ denom,
                               const float* __restrict__ v, float* __restrict__ agg, int E) {
    const int t = blockIdx.x * blockDim.x + threadIdx.x;
    if (t >= E * 8) return;
    const int e = t >> 3;
    const int c = t & 7;
    const int d0 = c * 8;
    const int hh = d0 >> 4;
    const int s = src[e], d = dst[e];
    const float w = ex[(size_t)e * HEADS + hh] /
                    (denom[(size_t)d * HEADS + hh] + 1e-9f);
    const float* vp = v + (size_t)s * 64 + d0;
    float* ap = agg + (size_t)d * 64 + d0;
#pragma unroll
    for (int j = 0; j < 8; ++j) atomicAdd(&ap[j], w * vp[j]);
}

// ---------------- gated residual + alpha mix (in-place x) ----------------
__global__ void gate_mix(const float* __restrict__ outb, float* __restrict__ x,
                         const float* __restrict__ x0,
                         const float* __restrict__ gw, const float* __restrict__ gb,
                         int N) {
    const int t = blockIdx.x * blockDim.x + threadIdx.x;
    if (t >= N) return;
    const float* o  = outb + (size_t)t * 64;
    float*       xp = x    + (size_t)t * 64;
    const float* zp = x0   + (size_t)t * 64;
    float s = gb[0];
#pragma unroll 8
    for (int d = 0; d < 64; ++d) {
        const float od = o[d], rd = xp[d];
        s += gw[d] * od + gw[64 + d] * rd + gw[128 + d] * (od - rd);
    }
    const float g = 1.f / (1.f + __expf(-s));
#pragma unroll 8
    for (int d = 0; d < 64; ++d) {
        const float mixed = g * o[d] + (1.f - g) * xp[d];
        xp[d] = ALPHA * mixed + (1.f - ALPHA) * zp[d];
    }
}

// =====================================================================
extern "C" void kernel_launch(void* const* d_in, const int* in_sizes, int n_in,
                              void* d_out, int out_size, void* d_ws, size_t ws_size,
                              hipStream_t stream) {
    const float* nodes    = (const float*)d_in[0];
    const int*   edge_idx = (const int*)  d_in[1];
    const float* emb_W    = (const float*)d_in[2];
    const float* emb_b    = (const float*)d_in[3];
    const float* ln_scale = (const float*)d_in[4];
    const float* ln_bias  = (const float*)d_in[5];
    const float* Wq       = (const float*)d_in[6];
    const float* Wk       = (const float*)d_in[7];
    const float* Wv       = (const float*)d_in[8];
    const float* Wo       = (const float*)d_in[9];
    const float* gate_W   = (const float*)d_in[10];
    const float* gate_b   = (const float*)d_in[11];

    const int N = in_sizes[0] / F_IN;
    const int E = in_sizes[1] / 2;
    const int* src = edge_idx;
    const int* dst = edge_idx + E;

    float* ws = (float*)d_ws;
    const size_t NF = (size_t)N * 64;
    float*    x0    = ws;
    float*    h     = ws + 1 * NF;
    float*    qb    = ws + 2 * NF;
    float*    kb    = ws + 3 * NF;
    float*    vb    = ws + 4 * NF;
    float*    agg   = ws + 5 * NF;
    float*    outb  = ws + 6 * NF;
    float*    elog  = ws + 7 * NF;                            // [E,4]
    unsigned* menc  = (unsigned*)(elog + (size_t)E * HEADS);  // [N,4]
    float*    denom = (float*)(menc + (size_t)N * HEADS);     // [N,4]

    float* x = (float*)d_out;

    const dim3 blkGemm(128);
    const dim3 grdGemm((N + MT - 1) / MT);
    const int  T = 256;

    // embedding: K=1433 (odd) -> unaligned A rows -> b32 NT async path
    gemm_wmma_n64<false><<<grdGemm, blkGemm, 0, stream>>>(nodes, emb_W, emb_b, x0, N, F_IN);
    hipMemcpyAsync(x, x0, NF * sizeof(float), hipMemcpyDeviceToDevice, stream);

    for (int l = 0; l < DEPTH; ++l) {
        const float* Wq_l = Wq + (size_t)l * 64 * 64;
        const float* Wk_l = Wk + (size_t)l * 64 * 64;
        const float* Wv_l = Wv + (size_t)l * 64 * 64;
        const float* Wo_l = Wo + (size_t)l * 64 * 64;

        layernorm_wave<<<(N * 32 + T - 1) / T, T, 0, stream>>>(
            x, h, ln_scale + l * 64, ln_bias + l * 64, N);

        gemm_wmma_n64<true><<<grdGemm, blkGemm, 0, stream>>>(h, Wq_l, nullptr, qb, N, 64);
        gemm_wmma_n64<true><<<grdGemm, blkGemm, 0, stream>>>(h, Wk_l, nullptr, kb, N, 64);
        gemm_wmma_n64<true><<<grdGemm, blkGemm, 0, stream>>>(h, Wv_l, nullptr, vb, N, 64);

        init_accum<<<(N * 64 + T - 1) / T, T, 0, stream>>>(agg, menc, denom, N);

        edge_logits<<<(E + T - 1) / T, T, 0, stream>>>(src, dst, qb, kb, elog, menc, E);
        edge_expsum<<<(E + T - 1) / T, T, 0, stream>>>(dst, elog, menc, denom, E);
        edge_aggregate<<<(E * 8 + T - 1) / T, T, 0, stream>>>(src, dst, elog, denom, vb, agg, E);

        gemm_wmma_n64<true><<<grdGemm, blkGemm, 0, stream>>>(agg, Wo_l, nullptr, outb, N, 64);

        gate_mix<<<(N + T - 1) / T, T, 0, stream>>>(
            outb, x, x0, gate_W + (size_t)l * 192, gate_b + l, N);
    }
}
